// Three_Sections_GNN_55688545960294
// MI455X (gfx1250) — matrix operand order repaired and checked by reading, not compiled, not run on hardware
//
#include <hip/hip_runtime.h>
#include <hip/hip_bf16.h>
#include <cmath>

typedef __attribute__((ext_vector_type(16))) _Float16 v16h;
typedef __attribute__((ext_vector_type(8)))  _Float16 v8h;
typedef __attribute__((ext_vector_type(8)))  float    v8f;

#define NNODES 10000
#define NE     160000
#define EFC    65
#define ET     (NE + NNODES)   // 170000 edges incl. self loops
#define LD     224             // padded feature leading dim (multiple of 32 floats -> 128B rows)
#define EC     32768           // edge chunk size
#define NGRAPH 32

// ---------------- A-operand fetch descriptor ----------------
struct AQuery {
  int mode;                 // 0: f32 plain (zero-padded rows), 1: f16 plain (zero-padded rows),
                            // 2: eav rows, 3: fcat rows
  const float*    A;  int lda;
  const _Float16* Ah;
  const float*    ea;       // edge_attr (E x 65)
  const float*    xi;       // node features (N x LD)
  const int*      srcF; const int* tgtF;
  int chunkStart;
};

__device__ __forceinline__ v8h pack8(float4 a, float4 b){
  v8h r;
  r[0]=(_Float16)a.x; r[1]=(_Float16)a.y; r[2]=(_Float16)a.z; r[3]=(_Float16)a.w;
  r[4]=(_Float16)b.x; r[5]=(_Float16)b.y; r[6]=(_Float16)b.z; r[7]=(_Float16)b.w;
  return r;
}

// ---------------- WMMA tiled GEMM: C = act(A@B + bias) * mask ----------------
// block = 256 threads = 8 waves; block tile 64(M) x 64(N); K-step 32 (f16 WMMA).
// LDS tiles stored pre-permuted into the CDNA5 WMMA fragment layout: one 32B
// vector LDS read per fragment, one 16B vector LDS store per staging thread.
// Modes 0/1 rely on zero-padded rows (Kp <= lda) -> branch-free vector staging.
// Out-of-range rows are clamped and discarded by the epilogue's row guard.
template <int MODE>
__global__ void __launch_bounds__(256) wmma_gemm_t(
    AQuery q, int M, int Kreal, int Ksteps,
    const _Float16* __restrict__ B, int ldb,
    const float* __restrict__ bias, int act, const float* __restrict__ maskRow,
    float* __restrict__ C, int ldc, _Float16* __restrict__ Ch, int ldch, int Nreal)
{
  __shared__ __align__(32) _Float16 As[64 * 32];   // permuted: [(mT*32+lane)*16 + e]
  __shared__ __align__(32) _Float16 Bs[64 * 32];   // permuted: [(nS*32+lane)*16 + e]
  const int tid  = threadIdx.x;
  const int lane = tid & 31;
  const int wave = tid >> 5;
  const int mT = wave & 3;          // 4 M-subtiles of 16
  const int nH = wave >> 2;         // 2 N-groups, each covering 2 N-subtiles
  const int bm0 = blockIdx.x * 64;
  const int bn0 = blockIdx.y * 64;
  const int h  = lane >> 4;
  const int ln = lane & 15;

  // ---- per-thread staging coordinates (constant across K loop) ----
  // A: thread owns 8 consecutive k of one row; inverse map h=(k>>3)&1, e=(k>>4)*8+(k&7)
  const int arow = tid >> 2;                 // 0..63
  const int ak0  = (tid & 3) * 8;            // 0,8,16,24
  _Float16* aDst = &As[(((arow >> 4) * 32) + ((ak0 >> 3) & 1) * 16 + (arow & 15)) * 16
                       + (ak0 >> 4) * 8];
  // B: thread owns 8 consecutive k of one column n' (h=(k>>3)&1, e=(k>>4)*8+(k&7))
  const int bn   = tid >> 2;                 // n' 0..63
  const int bk0  = (tid & 3) * 8;
  _Float16* bDst = &Bs[(((bn >> 4) * 32) + ((bk0 >> 3) & 1) * 16 + (bn & 15)) * 16 + (bk0 >> 4) * 8];
  const _Float16* __restrict__ pB = B + bn0 + bn;   // Np is multiple of 64: in-bounds

  // ---- row context for A fetch (hoisted gathers, clamped rows) ----
  const int  grow  = bm0 + arow;
  const int  rsafe = (grow < M) ? grow : 0;
  const float* pA = nullptr; const _Float16* pAh = nullptr;
  const float* ps = nullptr; const float* pt = nullptr; const float* pea = nullptr;
  bool inE = false;
  if constexpr (MODE == 0) { pA  = q.A  + (size_t)rsafe * q.lda; }
  else if constexpr (MODE == 1) { pAh = q.Ah + (size_t)rsafe * q.lda; }
  else {
    int e = q.chunkStart + rsafe;
    inE = e < NE;
    pea = q.ea + (size_t)(inE ? e : 0) * EFC;
    if constexpr (MODE == 3) {
      ps = q.xi + (size_t)q.srcF[e] * LD;
      pt = q.xi + (size_t)q.tgtF[e] * LD;
    }
  }

  v8f acc[2] = {};
  for (int kk = 0; kk < Ksteps; ++kk) {
    const int kb   = kk * 32;
    const int col0 = kb + ak0;                // multiple of 8; segment-uniform chunk
    v8h av = {};
    if constexpr (MODE == 0) {
      // Kp <= lda and pad columns are zero: unguarded, branch-free
      const float4* p = (const float4*)(pA + col0);   // rows 128B-aligned
      av = pack8(p[0], p[1]);
    } else if constexpr (MODE == 1) {
      av = *(const v8h*)(pAh + col0);                 // rows 16B-aligned, pads zero
    } else if constexpr (MODE == 2) {
      if (inE) {
        if (col0 + 7 < Kreal) {               // unaligned rows: unguarded scalar loads
#pragma unroll
          for (int i = 0; i < 8; ++i) av[i] = (_Float16)pea[col0 + i];
        } else if (col0 < Kreal) {            // tail chunk (col0=64): stay in-buffer
#pragma unroll
          for (int i = 0; i < 8; ++i) if (col0 + i < Kreal) av[i] = (_Float16)pea[col0 + i];
        }
      }
    } else {                                  // fcat: boundaries 200/400/600 chunk-aligned
      if (col0 + 7 < 200) {
        const float4* p = (const float4*)(ps + col0);
        av = pack8(p[0], p[1]);
      } else if (col0 >= 200 && col0 + 7 < 400) {
        const float4* p = (const float4*)(pt + (col0 - 200));
        av = pack8(p[0], p[1]);
      } else if (col0 >= 400 && col0 + 7 < 600) {
        const float4* a = (const float4*)(ps + (col0 - 400));
        const float4* b = (const float4*)(pt + (col0 - 400));
        float4 a0 = a[0], a1 = a[1], b0 = b[0], b1 = b[1];
        float4 d0 = make_float4(a0.x-b0.x, a0.y-b0.y, a0.z-b0.z, a0.w-b0.w);
        float4 d1 = make_float4(a1.x-b1.x, a1.y-b1.y, a1.z-b1.z, a1.w-b1.w);
        av = pack8(d0, d1);
      } else if (col0 >= 600 && inE) {        // eav tail (only col0=664 is mixed)
#pragma unroll
        for (int i = 0; i < 8; ++i) {
          int c = col0 + i - 600;
          if (c < EFC) av[i] = (_Float16)pea[c];
        }
      }
    }
    *reinterpret_cast<v8h*>(aDst) = av;
    v8h bv;
#pragma unroll
    for (int i = 0; i < 8; ++i)
      bv[i] = pB[(size_t)(kb + bk0 + i) * ldb];
    *reinterpret_cast<v8h*>(bDst) = bv;
    __syncthreads();
    v16h af  = *reinterpret_cast<const v16h*>(&As[(mT * 32 + lane) * 16]);
    v16h bf0 = *reinterpret_cast<const v16h*>(&Bs[((nH * 2 + 0) * 32 + lane) * 16]);
    v16h bf1 = *reinterpret_cast<const v16h*>(&Bs[((nH * 2 + 1) * 32 + lane) * 16]);
    acc[0] = __builtin_amdgcn_wmma_f32_16x16x32_f16(false, af, false, bf0, (short)0, acc[0], false, false);
    acc[1] = __builtin_amdgcn_wmma_f32_16x16x32_f16(false, af, false, bf1, (short)0, acc[1], false, false);
    __syncthreads();
  }
#pragma unroll
  for (int j = 0; j < 2; ++j) {
    int col = bn0 + (nH * 2 + j) * 16 + ln;
#pragma unroll
    for (int v = 0; v < 8; ++v) {
      int row = bm0 + mT * 16 + h * 8 + v;     // C/D layout: M = v + half*8, N = lane&15
      if (row < M && col < Nreal) {
        float val = acc[j][v];
        if (bias) val += bias[col];
        if (act == 1) val = val > 0.0f ? val : (expf(val) - 1.0f);   // ELU
        if (maskRow) val *= maskRow[row];
        if (C)  C [(size_t)row * ldc  + col] = val;
        if (Ch) Ch[(size_t)row * ldch + col] = (_Float16)val;
      }
    }
  }
}

// ---------------- small helper kernels ----------------
__global__ void k_zero(float* p, int n){ int i = blockIdx.x*256 + threadIdx.x; if (i < n) p[i] = 0.0f; }
__global__ void k_zero32(unsigned* p, int n){ int i = blockIdx.x*256 + threadIdx.x; if (i < n) p[i] = 0u; }
__global__ void k_copy(const float* s, float* d, int n){ int i = blockIdx.x*256 + threadIdx.x; if (i < n) d[i] = s[i]; }

__global__ void k_pad_x(const float* x, float* xA, int n){
  int i = blockIdx.x*256 + threadIdx.x; if (i >= n) return;
  int r = i / LD, c = i % LD;
  xA[i] = (c < 92) ? x[(size_t)r * 92 + c] : 0.0f;
}
__global__ void k_edges(const int* ei, int* srcF, int* tgtF, int n){
  int i = blockIdx.x*256 + threadIdx.x; if (i >= n) return;
  srcF[i] = (i < NE) ? ei[i]      : (i - NE);
  tgtF[i] = (i < NE) ? ei[NE + i] : (i - NE);
}
__global__ void k_mask(const float* ea, float* m, int n){
  int i = blockIdx.x*256 + threadIdx.x; if (i >= n) return;
  m[i] = (i < NE) ? ((ea[(size_t)i * EFC] < 8.0f) ? 1.0f : 0.0f) : 1.0f;  // self-loop ea=0
}
__global__ void k_cvtW(const float* W, _Float16* Wh, int K, int N, int Kp, int Np, int n){
  int i = blockIdx.x*256 + threadIdx.x; if (i >= n) return;
  int k = i / Np, c = i % Np;
  Wh[i] = (k < K && c < N) ? (_Float16)W[(size_t)k * N + c] : (_Float16)0.0f;
}
// Mt[i][j] = sum_c kW[i,c] * (aW[c]*scale) * qW[j,c]   (B operand for t = f @ Mt)
__global__ void k_buildMt(const float* qW, const float* kW, const float* aW, float scale,
                          _Float16* Mt, int d, int Np, int n){
  int i = blockIdx.x*256 + threadIdx.x; if (i >= n) return;
  int r = i / Np, c = i % Np;
  float s = 0.0f;
  if (r < d && c < d) {
    for (int t = 0; t < d; ++t) s += kW[(size_t)r * d + t] * aW[t] * qW[(size_t)c * d + t];
    s *= scale;
  }
  Mt[i] = (_Float16)s;
}
__global__ void k_conv_scatter(const float* edgeC, const float* xold,
                               const int* srcF, const int* tgtF, int cs, int rows, int tin,
                               float* acc, int n){
  int i = blockIdx.x*256 + threadIdx.x; if (i >= n) return;
  int r = i / tin, c = i % tin;
  int e = cs + r;
  float m = edgeC[(size_t)r * LD + c] * xold[(size_t)tgtF[e] * LD + c];   // mask already in edgeC
  atomicAdd(&acc[(size_t)srcF[e] * LD + c], m);
}
__global__ void k_bn_stats(const float* h, int nrow, float* mu, float* inv){
  __shared__ float s1[256], s2[256];
  int c = blockIdx.x;
  float a = 0.0f, b = 0.0f;
  for (int r = threadIdx.x; r < nrow; r += 256) { float v = h[(size_t)r * LD + c]; a += v; b += v * v; }
  s1[threadIdx.x] = a; s2[threadIdx.x] = b; __syncthreads();
  for (int st = 128; st > 0; st >>= 1) {
    if (threadIdx.x < st) { s1[threadIdx.x] += s1[threadIdx.x + st]; s2[threadIdx.x] += s2[threadIdx.x + st]; }
    __syncthreads();
  }
  if (threadIdx.x == 0) {
    float m = s1[0] / nrow;
    float var = s2[0] / nrow - m * m;
    mu[c] = m; inv[c] = rsqrtf(var + 1e-5f);
  }
}
__global__ void k_bn_apply(const float* h, const float* mu, const float* inv,
                           const float* g, const float* b, float* out, int d, int n){
  int i = blockIdx.x*256 + threadIdx.x; if (i >= n) return;
  int r = i / d, c = i % d;
  out[(size_t)r * LD + c] = (h[(size_t)r * LD + c] - mu[c]) * inv[c] * g[c] + b[c];
}
__global__ void k_att_a(const float* tC, const float* xi, const int* srcF,
                        int cs, int rows, float* a, float* segA){
  int i = blockIdx.x*256 + threadIdx.x; if (i >= rows) return;
  int e = cs + i, s = srcF[e];
  const float4* t = (const float4*)(tC + (size_t)i * LD);   // 128B-aligned rows
  const float4* x = (const float4*)(xi + (size_t)s * LD);
  float dot = 0.0f;
#pragma unroll 2
  for (int c = 0; c < 50; ++c) {
    float4 tv = t[c], xv = x[c];
    dot += xv.x * tv.x + xv.y * tv.y + xv.z * tv.z + xv.w * tv.w;
  }
  float av = tanhf(dot);
  a[e] = av;
  atomicAdd(&segA[s], av);
}
__global__ void k_att_s(const float* a, const float* segA, const int* srcF, float* s, int n){
  int i = blockIdx.x*256 + threadIdx.x; if (i >= n) return;
  s[i] = expf(a[i] - segA[srcF[i]]);
}
__global__ void k_att_scatter(const _Float16* fF, const float* s, const int* srcF, float* aggr, int n){
  int i = blockIdx.x*256 + threadIdx.x; if (i >= n) return;
  int e = i / 200, c = i % 200;
  atomicAdd(&aggr[(size_t)srcF[e] * LD + c], s[e] * (float)fF[(size_t)e * LD + c]);
}
__global__ void k_add(const float* x, const float* y, float* out, int n){
  int i = blockIdx.x*256 + threadIdx.x; if (i >= n) return;
  int r = i / 200, c = i % 200;
  size_t o = (size_t)r * LD + c;
  out[o] = x[o] + y[o];
}
__global__ void k_pool_cnt(const int* batch, float* cnt, int n){
  int i = blockIdx.x*256 + threadIdx.x; if (i >= n) return;
  atomicAdd(&cnt[batch[i]], 1.0f);
}
__global__ void k_pool_sum(const float* o0, const float* o1, const float* o2,
                           const int* batch, float* gsum, int n){
  int i = blockIdx.x*256 + threadIdx.x; if (i >= n) return;
  int nd = i / 600, c = i % 600;
  int hh = c / 200, cc = c % 200;
  const float* o = (hh == 0) ? o0 : ((hh == 1) ? o1 : o2);
  atomicAdd(&gsum[(size_t)batch[nd] * 600 + c], o[(size_t)nd * LD + cc]);
}
__global__ void k_post_mid(const float* gsum, const float* cnt, const float* W, const float* b,
                           const float* prelu, float* pm, int n){
  int i = blockIdx.x*256 + threadIdx.x; if (i >= n) return;
  int g = i / 100, j = i % 100;
  float ic = 1.0f / cnt[g];
  float h = b[j];
  for (int k = 0; k < 600; ++k) h += gsum[(size_t)g * 600 + k] * ic * W[(size_t)k * 100 + j];
  float pr = *prelu;
  pm[i] = (h >= 0.0f) ? h : pr * h;
}
__global__ void k_post_last(const float* pm, const float* W, const float* b, float* out, int n){
  int i = blockIdx.x*256 + threadIdx.x; if (i >= n) return;
  int g = i / 100, j = i % 100;
  float h = b[j];
  for (int k = 0; k < 100; ++k) h += pm[(size_t)g * 100 + k] * W[(size_t)k * 100 + j];
  out[i] = h;
}

// ---------------- host helpers ----------------
static inline AQuery qF32(const float* A, int lda){ AQuery q{}; q.mode=0; q.A=A; q.lda=lda; return q; }
static inline AQuery qF16(const _Float16* A, int lda){ AQuery q{}; q.mode=1; q.Ah=A; q.lda=lda; return q; }
static inline AQuery qEAV(const float* ea, int cs){ AQuery q{}; q.mode=2; q.ea=ea; q.chunkStart=cs; return q; }
static inline AQuery qFCAT(const float* xi, const float* ea, const int* s, const int* t){
  AQuery q{}; q.mode=3; q.xi=xi; q.ea=ea; q.srcF=s; q.tgtF=t; return q; }

static inline int np64(int n){ return ((n + 63) / 64) * 64; }

static inline void launch_gemm(hipStream_t st, AQuery q, int M, int Kreal,
                               const _Float16* B, int ldb, const float* bias, int act,
                               const float* maskRow, float* C, int ldc,
                               _Float16* Ch, int ldch, int Nreal){
  int Ksteps = (Kreal + 31) / 32;
  dim3 grid((M + 63) / 64, (Nreal + 63) / 64);
  switch (q.mode) {
    case 0: wmma_gemm_t<0><<<grid, 256, 0, st>>>(q, M, Kreal, Ksteps, B, ldb, bias, act, maskRow, C, ldc, Ch, ldch, Nreal); break;
    case 1: wmma_gemm_t<1><<<grid, 256, 0, st>>>(q, M, Kreal, Ksteps, B, ldb, bias, act, maskRow, C, ldc, Ch, ldch, Nreal); break;
    case 2: wmma_gemm_t<2><<<grid, 256, 0, st>>>(q, M, Kreal, Ksteps, B, ldb, bias, act, maskRow, C, ldc, Ch, ldch, Nreal); break;
    default: wmma_gemm_t<3><<<grid, 256, 0, st>>>(q, M, Kreal, Ksteps, B, ldb, bias, act, maskRow, C, ldc, Ch, ldch, Nreal); break;
  }
}

#define FLAT(kern, n, ...) kern<<<dim3(((n) + 255) / 256), dim3(256), 0, stream>>>(__VA_ARGS__, (n))

extern "C" void kernel_launch(void* const* d_in, const int* in_sizes, int n_in,
                              void* d_out, int out_size, void* d_ws, size_t ws_size,
                              hipStream_t stream) {
  (void)in_sizes; (void)n_in; (void)out_size; (void)ws_size;
  const float* x_in = (const float*)d_in[0];
  const int*   ei   = (const int*)  d_in[1];
  const float* ea   = (const float*)d_in[2];
  const int*   batch= (const int*)  d_in[3];
  // d_in[4] = n_graphs (constant 32)

  // ---- workspace carve ----
  size_t off = 0;
  auto carve = [&](size_t bytes) -> void* {
    void* p = (char*)d_ws + off;
    off = (off + bytes + 255) & ~(size_t)255;
    return p;
  };
  float* xA    = (float*)carve((size_t)NNODES * LD * 4);
  float* xAcc  = (float*)carve((size_t)NNODES * LD * 4);
  float* hbuf  = (float*)carve((size_t)NNODES * LD * 4);
  float* aggr  = (float*)carve((size_t)NNODES * LD * 4);
  float* oH[3];
  for (int i = 0; i < 3; ++i) oH[i] = (float*)carve((size_t)NNODES * LD * 4);
  float* edgeChunk = (float*)carve((size_t)EC * LD * 4);
  _Float16* fFull  = (_Float16*)carve((size_t)ET * LD * 2);
  float* maskBuf = (float*)carve((size_t)ET * 4);
  float* aBuf    = (float*)carve((size_t)ET * 4);
  float* sBuf    = (float*)carve((size_t)ET * 4);
  float* segA    = (float*)carve((size_t)NNODES * 4);
  int*   srcF    = (int*)carve((size_t)ET * 4);
  int*   tgtF    = (int*)carve((size_t)ET * 4);
  _Float16* wslot1 = (_Float16*)carve((size_t)672 * 256 * 2);
  _Float16* wslot2 = (_Float16*)carve((size_t)224 * 256 * 2);
  float* mu   = (float*)carve(256 * 4);
  float* inv  = (float*)carve(256 * 4);
  float* gsum = (float*)carve((size_t)NGRAPH * 600 * 4);
  float* cnt  = (float*)carve(NGRAPH * 4);
  float* pm   = (float*)carve((size_t)NGRAPH * 100 * 4);

  // ---- init ----
  FLAT(k_pad_x, NNODES * LD, x_in, xA);
  FLAT(k_edges, ET, ei, srcF, tgtF);
  FLAT(k_mask,  ET, ea, maskBuf);
  FLAT(k_zero32, ET * LD / 2, (unsigned*)fFull);   // zero pads so mode-1 staging is unguarded

  // ---- conv layers ----
  int tin = 92;
  const int dims0[3] = {200, 200, 200};
  for (int l = 0; l < 3; ++l) {
    const float* edge_W = (const float*)d_in[5 + l * 6 + 0];
    const float* edge_b = (const float*)d_in[5 + l * 6 + 1];
    const float* upd_W  = (const float*)d_in[5 + l * 6 + 2];
    const float* upd_b  = (const float*)d_in[5 + l * 6 + 3];
    const float* bn_g   = (const float*)d_in[5 + l * 6 + 4];
    const float* bn_b   = (const float*)d_in[5 + l * 6 + 5];
    int d = dims0[l];
    int NpT = np64(tin);
    FLAT(k_cvtW, 96 * NpT, edge_W, wslot1, EFC, tin, 96, NpT);
    FLAT(k_copy, NNODES * LD, xA, xAcc);
    for (int cs = 0; cs < ET; cs += EC) {
      int rows = (ET - cs < EC) ? (ET - cs) : EC;
      // edge = elu(eav @ edge_W + b) * mask   (mask fused in epilogue)
      launch_gemm(stream, qEAV(ea, cs), rows, EFC, wslot1, NpT, edge_b, 1,
                  maskBuf + cs, edgeChunk, LD, nullptr, 0, tin);
      FLAT(k_conv_scatter, rows * tin, edgeChunk, xA, srcF, tgtF, cs, rows, tin, xAcc);
    }
    int KpU = ((tin + 31) / 32) * 32;
    FLAT(k_cvtW, KpU * 256, upd_W, wslot1, tin, d, KpU, 256);
    launch_gemm(stream, qF32(xAcc, LD), NNODES, tin, wslot1, 256, upd_b, 0,
                nullptr, hbuf, LD, nullptr, 0, d);
    k_bn_stats<<<dim3(d), dim3(256), 0, stream>>>(hbuf, NNODES, mu, inv);
    FLAT(k_bn_apply, NNODES * d, hbuf, mu, inv, bn_g, bn_b, xA, d);
    tin = d;
  }

  // ---- attention layer (3 heads, xi = xA for all) ----
  float scale = 1.0f / sqrtf(200.0f);
  for (int hh = 0; hh < 3; ++hh) {
    int base = 23 + hh * 9;
    const float* f_W   = (const float*)d_in[base + 0];
    const float* f_b   = (const float*)d_in[base + 1];
    const float* q_W   = (const float*)d_in[base + 2];
    const float* k_W   = (const float*)d_in[base + 3];
    const float* a_W   = (const float*)d_in[base + 4];
    const float* upd_W = (const float*)d_in[base + 5];
    const float* upd_b = (const float*)d_in[base + 6];
    const float* bn_g  = (const float*)d_in[base + 7];
    const float* bn_b  = (const float*)d_in[base + 8];

    // f = elu([ni,nj,ni-nj,eav] @ f_W + f_b) * mask  -> fFull (f16), fcat built in LDS
    FLAT(k_cvtW, 672 * 256, f_W, wslot1, 665, 200, 672, 256);
    launch_gemm(stream, qFCAT(xA, ea, srcF, tgtF), ET, 665, wslot1, 256, f_b, 1,
                maskBuf, nullptr, 0, fFull, LD, 200);

    // Mt = kW * diag(aW*scale) * qW^T  (so t = f @ Mt, a = tanh(ni . t))
    FLAT(k_buildMt, 224 * 256, q_W, k_W, a_W, scale, wslot2, 200, 256);
    FLAT(k_zero, NNODES, segA);
    for (int cs = 0; cs < ET; cs += EC) {
      int rows = (ET - cs < EC) ? (ET - cs) : EC;
      launch_gemm(stream, qF16(fFull + (size_t)cs * LD, LD), rows, 200, wslot2, 256,
                  nullptr, 0, nullptr, edgeChunk, LD, nullptr, 0, 200);
      k_att_a<<<dim3((rows + 255) / 256), dim3(256), 0, stream>>>(edgeChunk, xA, srcF, cs, rows, aBuf, segA);
    }
    FLAT(k_att_s, ET, aBuf, segA, srcF, sBuf);
    FLAT(k_zero, NNODES * LD, aggr);
    FLAT(k_att_scatter, ET * 200, fFull, sBuf, srcF, aggr);
    FLAT(k_add, NNODES * 200, xA, aggr, xAcc);
    FLAT(k_cvtW, 224 * 256, upd_W, wslot1, 200, 200, 224, 256);
    launch_gemm(stream, qF32(xAcc, LD), NNODES, 200, wslot1, 256, upd_b, 0,
                nullptr, hbuf, LD, nullptr, 0, 200);
    k_bn_stats<<<dim3(200), dim3(256), 0, stream>>>(hbuf, NNODES, mu, inv);
    FLAT(k_bn_apply, NNODES * 200, hbuf, mu, inv, bn_g, bn_b, oH[hh], 200);
  }

  // ---- pooling + post MLP ----
  FLAT(k_zero, NGRAPH * 600, gsum);
  FLAT(k_zero, NGRAPH, cnt);
  FLAT(k_pool_cnt, NNODES, batch, cnt);
  FLAT(k_pool_sum, NNODES * 600, oH[0], oH[1], oH[2], batch, gsum);
  const float* W1 = (const float*)d_in[50];
  const float* b1 = (const float*)d_in[51];
  const float* pr = (const float*)d_in[52];
  const float* W2 = (const float*)d_in[53];
  const float* b2 = (const float*)d_in[54];
  FLAT(k_post_mid,  NGRAPH * 100, gsum, cnt, W1, b1, pr, pm);
  FLAT(k_post_last, NGRAPH * 100, pm, W2, b2, (float*)d_out);
}
